// MV_LSTM_SUPERVISED_52501680226917
// MI455X (gfx1250) — matrix operand order repaired and checked by reading, not compiled, not run on hardware
//
#include <hip/hip_runtime.h>

// MV_LSTM_SUPERVISED fused CDNA5 implementation.
// Pipeline: pack weights (f32->f16, WMMA layouts) -> combine GEMM (WMMA) ->
// fully fused 2-layer LSTM recurrence + output dot (WMMA per timestep).

typedef __attribute__((ext_vector_type(16))) _Float16 v16h;
typedef __attribute__((ext_vector_type(8)))  float    v8f;

#define B_   4096
#define S_   128
#define F_   8
#define SF   1024      // S_*F_
#define H1_  64
#define H2_  32
#define N1_  256       // 4*H1
#define N2_  128       // 4*H2
#define KPAD 96        // padded K for both LSTM gate GEMMs
#define G1LD 260       // padded LDS stride for gates1 (bank-conflict free)
#define G2LD 132       // padded LDS stride for gates2

// gfx1250 has a hardware tanh transcendental (v_tanh_f32).
#if __has_builtin(__builtin_amdgcn_tanhf)
#define TANHF_(x) __builtin_amdgcn_tanhf(x)
#else
#define TANHF_(x) tanhf(x)
#endif

__device__ __forceinline__ float fsig(float x) {
  // sigmoid(x) = 0.5*tanh(x/2) + 0.5  -> single TRANS op + FMA on gfx1250
  return __builtin_fmaf(0.5f, TANHF_(0.5f * x), 0.5f);
}

// ---- WMMA fragment loaders (CDNA5 ISA 7.12.2 layouts, wave32) -------------

// A-matrix 16x32 f16 fragment from row-major halfs u[16][ldu], K offset kbase.
// VGPR pair p, lane group g: K = (p<4 ? 2p : 16+2(p-4)) + 8g ; M = lane&15.
__device__ __forceinline__ v16h load_a_frag(const _Float16* u, int ldu,
                                            int kbase, int lane) {
  const int m = lane & 15;
  const int g = (lane >> 4) & 1;
  v16h a;
#pragma unroll
  for (int p = 0; p < 8; ++p) {
    const int k = kbase + ((p < 4) ? (2 * p) : (16 + 2 * (p - 4))) + 8 * g;
    a[2 * p]     = u[m * ldu + k];
    a[2 * p + 1] = u[m * ldu + k + 1];
  }
  return a;
}

// B-matrix 32x16 f16 fragment. Source W is row-major [N][ldk] (i.e. W^T of the
// math B), so element (k,n) = W[n][k]. Lane = N, lane group g: K = (p<4 ? 2p : 16+2(p-4)) + 8g + e.
__device__ __forceinline__ v16h load_b_frag(const _Float16* W, int ldk,
                                            int n0, int kbase, int lane) {
  const int n = n0 + (lane & 15);
  const int g = (lane >> 4) & 1;
  v16h b;
#pragma unroll
  for (int p = 0; p < 8; ++p) {
    const int k = kbase + ((p < 4) ? (2 * p) : (16 + 2 * (p - 4))) + 8 * g;
    b[2 * p]     = W[n * ldk + k];
    b[2 * p + 1] = W[n * ldk + k + 1];
  }
  return b;
}

// ---- weight packing kernels ------------------------------------------------

// W_combine f32 [1024][1025] -> Wc16 f16 [1024][1024] + time column f32 [1024]
__global__ void k_pack_wcombine(const float* __restrict__ Wc,
                                _Float16* __restrict__ Wc16,
                                float* __restrict__ tcol) {
  int i = blockIdx.x * blockDim.x + threadIdx.x;
  if (i < SF * SF) {
    int n = i >> 10, k = i & 1023;
    Wc16[i] = (_Float16)Wc[(size_t)n * (SF + 1) + k];
    if (k == 0) tcol[n] = Wc[(size_t)n * (SF + 1) + SF];
  }
}

// Wc1 f16 [256][96] = [ W_ih1[256][8] | W_hh1[256][64] | zero pad[24] ]
__global__ void k_pack_lstm1(const float* __restrict__ Wih,
                             const float* __restrict__ Whh,
                             _Float16* __restrict__ Wc1) {
  int i = blockIdx.x * blockDim.x + threadIdx.x;
  if (i < N1_ * KPAD) {
    int n = i / KPAD, k = i % KPAD;
    float v = (k < F_) ? Wih[n * F_ + k]
            : (k < F_ + H1_) ? Whh[n * H1_ + (k - F_)] : 0.0f;
    Wc1[i] = (_Float16)v;
  }
}

// Wc2 f16 [128][96] = [ W_ih2[128][64] | W_hh2[128][32] ]; bias2 = b_ih2+b_hh2
__global__ void k_pack_lstm2(const float* __restrict__ Wih,
                             const float* __restrict__ Whh,
                             const float* __restrict__ bih,
                             const float* __restrict__ bhh,
                             _Float16* __restrict__ Wc2,
                             float* __restrict__ bias2) {
  int i = blockIdx.x * blockDim.x + threadIdx.x;
  if (i < N2_ * KPAD) {
    int n = i / KPAD, k = i % KPAD;
    float v = (k < H1_) ? Wih[n * H1_ + k] : Whh[n * H2_ + (k - H1_)];
    Wc2[i] = (_Float16)v;
  }
  if (i < N2_) bias2[i] = bih[i] + bhh[i];
}

// ---- combine: xf16 = relu(concat(x,time) @ W_combine^T + b) ---------------
// grid (B/16, 1024/64), 128 threads = 4 waves; wave w owns one 16-col tile.
__global__ void __launch_bounds__(128) k_combine(
    const float* __restrict__ x,        // [B,1024] f32
    const float* __restrict__ timev,    // [B]
    const float* __restrict__ bcomb,    // [1024]
    const _Float16* __restrict__ Wc16,  // [1024][1024]
    const float* __restrict__ tcol,     // [1024]
    _Float16* __restrict__ xf16) {      // [B,1024]
  __shared__ __align__(16) _Float16 As[16][32];
  const int tid  = threadIdx.x;
  const int lane = tid & 31;
  const int w    = tid >> 5;
  const int b0   = blockIdx.x * 16;
  const int n0   = blockIdx.y * 64 + w * 16;

  v8f acc = {0.f, 0.f, 0.f, 0.f, 0.f, 0.f, 0.f, 0.f};
  for (int kc = 0; kc < 32; ++kc) {
    const int kb = kc * 32;
    // stage 16x32 A tile to LDS (f32 -> f16), 4 elems/thread
    int idx = tid * 4;
#pragma unroll
    for (int q = 0; q < 4; ++q, ++idx) {
      int m = idx >> 5, kk = idx & 31;
      As[m][kk] = (_Float16)x[(size_t)(b0 + m) * SF + kb + kk];
    }
    __syncthreads();
    v16h a = load_a_frag(&As[0][0], 32, 0, lane);
    v16h b = load_b_frag(Wc16, SF, n0, kb, lane);
    acc = __builtin_amdgcn_wmma_f32_16x16x32_f16(false, a, false, b,
                                                 (short)0, acc, false, false);
    __syncthreads();
  }
  const int n = n0 + (lane & 15);
  const int g = (lane >> 4) & 1;
  const float tc = tcol[n], bb = bcomb[n];
#pragma unroll
  for (int r = 0; r < 8; ++r) {
    const int m = b0 + r + 8 * g;
    float v = acc[r] + timev[m] * tc + bb;
    v = v > 0.f ? v : 0.f;
    xf16[(size_t)m * SF + n] = (_Float16)v;
  }
}

// ---- fused 2-layer LSTM + output projection -------------------------------
// grid B/16, 256 threads = 8 waves. Wave w owns:
//   layer1 gate cols [32w, 32w+32)  (2 WMMA N-tiles)
//   layer2 gate cols [16w, 16w+16)  (1 WMMA N-tile)
// u1 = [x_t(8) | h1(64) | pad(24)] f16 ; u2 = [h1_t(64) | h2(32)] f16.
__global__ void __launch_bounds__(256) k_lstm_fused(
    const _Float16* __restrict__ xf16,  // [B,1024]
    const _Float16* __restrict__ Wc1,   // [256][96]
    const _Float16* __restrict__ Wc2,   // [128][96]
    const float* __restrict__ bias2,    // [128]
    const float* __restrict__ Wout,     // [4096]  (index t*32+j)
    const float* __restrict__ bout,     // [1]
    float* __restrict__ out) {          // [B]
  __shared__ __align__(16) _Float16 u1[16][KPAD];
  __shared__ __align__(16) _Float16 u2[16][KPAD];
  __shared__ float g1s[16][G1LD];
  __shared__ float g2s[16][G2LD];
  __shared__ float red[256];

  const int tid  = threadIdx.x;
  const int lane = tid & 31;
  const int w    = tid >> 5;
  const int b0   = blockIdx.x * 16;

  // zero h/c staging (covers zero initial states + K padding)
  for (int i = tid; i < 16 * KPAD; i += 256) {
    (&u1[0][0])[i] = (_Float16)0.f;
    (&u2[0][0])[i] = (_Float16)0.f;
  }

  // preload time-invariant weight fragments into registers
  v16h b1f[2][3], b2f[3];
#pragma unroll
  for (int nt = 0; nt < 2; ++nt)
#pragma unroll
    for (int kc = 0; kc < 3; ++kc)
      b1f[nt][kc] = load_b_frag(Wc1, KPAD, 32 * w + 16 * nt, 32 * kc, lane);
#pragma unroll
  for (int kc = 0; kc < 3; ++kc)
    b2f[kc] = load_b_frag(Wc2, KPAD, 16 * w, 32 * kc, lane);

  v8f cb2;  // layer2 bias as C-fragment (depends only on N)
  {
    float bv = bias2[16 * w + (lane & 15)];
#pragma unroll
    for (int r = 0; r < 8; ++r) cb2[r] = bv;
  }

  // per-thread recurrent cell state: thread owns batch eb, 4 L1 / 2 L2 units
  const int eb = tid >> 4;
  const int j1 = (tid & 15) * 4;
  const int j2 = (tid & 15) * 2;
  float c1r[4], c2r[2];
#pragma unroll
  for (int q = 0; q < 4; ++q) c1r[q] = 0.f;
#pragma unroll
  for (int q = 0; q < 2; ++q) c2r[q] = 0.f;
  float outacc = 0.f;

  __syncthreads();

#pragma unroll 1
  for (int t = 0; t < S_; ++t) {
    // stage x_t (16x8 f16) into u1[:,0:8]
    if (tid < 128)
      u1[tid >> 3][tid & 7] =
          xf16[(size_t)(b0 + (tid >> 3)) * SF + t * F_ + (tid & 7)];
    __syncthreads();

    // GEMM1: gates1(16x256) = u1(16x96) @ Wc1^T
    {
      v16h a0 = load_a_frag(&u1[0][0], KPAD, 0, lane);
      v16h a1 = load_a_frag(&u1[0][0], KPAD, 32, lane);
      v16h a2 = load_a_frag(&u1[0][0], KPAD, 64, lane);
#pragma unroll
      for (int nt = 0; nt < 2; ++nt) {
        v8f acc = {0.f, 0.f, 0.f, 0.f, 0.f, 0.f, 0.f, 0.f};
        acc = __builtin_amdgcn_wmma_f32_16x16x32_f16(false, a0, false, b1f[nt][0], (short)0, acc, false, false);
        acc = __builtin_amdgcn_wmma_f32_16x16x32_f16(false, a1, false, b1f[nt][1], (short)0, acc, false, false);
        acc = __builtin_amdgcn_wmma_f32_16x16x32_f16(false, a2, false, b1f[nt][2], (short)0, acc, false, false);
        const int n = 32 * w + 16 * nt + (lane & 15);
        const int g = (lane >> 4) & 1;
#pragma unroll
        for (int r = 0; r < 8; ++r) g1s[r + 8 * g][n] = acc[r];
      }
    }
    __syncthreads();

    // layer1 elementwise: c,h update ; h1_t -> u1[:,8:72] and u2[:,0:64]
#pragma unroll
    for (int q = 0; q < 4; ++q) {
      const int j = j1 + q;
      float iv = g1s[eb][j], fv = g1s[eb][64 + j];
      float gv = g1s[eb][128 + j], ov = g1s[eb][192 + j];
      float c = fsig(fv) * c1r[q] + fsig(iv) * TANHF_(gv);
      c1r[q] = c;
      _Float16 hh = (_Float16)(fsig(ov) * TANHF_(c));
      u1[eb][8 + j] = hh;
      u2[eb][j]     = hh;
    }
    __syncthreads();

    // GEMM2: gates2(16x128) = u2(16x96) @ Wc2^T + bias2
    {
      v16h a0 = load_a_frag(&u2[0][0], KPAD, 0, lane);
      v16h a1 = load_a_frag(&u2[0][0], KPAD, 32, lane);
      v16h a2 = load_a_frag(&u2[0][0], KPAD, 64, lane);
      v8f acc = cb2;
      acc = __builtin_amdgcn_wmma_f32_16x16x32_f16(false, a0, false, b2f[0], (short)0, acc, false, false);
      acc = __builtin_amdgcn_wmma_f32_16x16x32_f16(false, a1, false, b2f[1], (short)0, acc, false, false);
      acc = __builtin_amdgcn_wmma_f32_16x16x32_f16(false, a2, false, b2f[2], (short)0, acc, false, false);
      const int n = 16 * w + (lane & 15);
      const int g = (lane >> 4) & 1;
#pragma unroll
      for (int r = 0; r < 8; ++r) g2s[r + 8 * g][n] = acc[r];
    }
    __syncthreads();

    // layer2 elementwise + fused output dot accumulation
#pragma unroll
    for (int q = 0; q < 2; ++q) {
      const int j = j2 + q;
      float iv = g2s[eb][j], fv = g2s[eb][32 + j];
      float gv = g2s[eb][64 + j], ov = g2s[eb][96 + j];
      float c = fsig(fv) * c2r[q] + fsig(iv) * TANHF_(gv);
      c2r[q] = c;
      float h = fsig(ov) * TANHF_(c);
      u2[eb][64 + j] = (_Float16)h;
      outacc += Wout[t * H2_ + j] * h;
    }
    __syncthreads();
  }

  // reduce 16 partial sums per batch row
  red[tid] = outacc;
  __syncthreads();
  if (tid < 16) {
    float s = 0.f;
#pragma unroll
    for (int q = 0; q < 16; ++q) s += red[tid * 16 + q];
    out[b0 + tid] = s + bout[0];
  }
}

// ---- launcher --------------------------------------------------------------

extern "C" void kernel_launch(void* const* d_in, const int* in_sizes, int n_in,
                              void* d_out, int out_size, void* d_ws,
                              size_t ws_size, hipStream_t stream) {
  (void)in_sizes; (void)n_in; (void)out_size; (void)ws_size;
  const float* x     = (const float*)d_in[0];
  const float* timev = (const float*)d_in[1];
  const float* Wcomb = (const float*)d_in[2];
  const float* bcomb = (const float*)d_in[3];
  const float* Wih1  = (const float*)d_in[4];
  const float* Whh1  = (const float*)d_in[5];
  const float* Wih2  = (const float*)d_in[6];
  const float* Whh2  = (const float*)d_in[7];
  const float* bih2  = (const float*)d_in[8];
  const float* bhh2  = (const float*)d_in[9];
  const float* Wout  = (const float*)d_in[10];
  const float* bout  = (const float*)d_in[11];
  float* out = (float*)d_out;

  // workspace carve-up (~10.6 MB total)
  char* ws = (char*)d_ws;
  _Float16* Wc16  = (_Float16*)ws; ws += (size_t)SF * SF * sizeof(_Float16);
  float*    tcol  = (float*)ws;    ws += (size_t)SF * sizeof(float);
  _Float16* xf16  = (_Float16*)ws; ws += (size_t)B_ * SF * sizeof(_Float16);
  _Float16* Wc1   = (_Float16*)ws; ws += (size_t)N1_ * KPAD * sizeof(_Float16);
  _Float16* Wc2   = (_Float16*)ws; ws += (size_t)N2_ * KPAD * sizeof(_Float16);
  float*    bias2 = (float*)ws;    ws += (size_t)N2_ * sizeof(float);

  k_pack_wcombine<<<(SF * SF + 255) / 256, 256, 0, stream>>>(Wcomb, Wc16, tcol);
  k_pack_lstm1<<<(N1_ * KPAD + 255) / 256, 256, 0, stream>>>(Wih1, Whh1, Wc1);
  k_pack_lstm2<<<(N2_ * KPAD + 255) / 256, 256, 0, stream>>>(Wih2, Whh2, bih2,
                                                             bhh2, Wc2, bias2);
  dim3 gc(B_ / 16, SF / 64);
  k_combine<<<gc, 128, 0, stream>>>(x, timev, bcomb, Wc16, tcol, xf16);
  k_lstm_fused<<<B_ / 16, 256, 0, stream>>>(xf16, Wc1, Wc2, bias2, Wout, bout,
                                            out);
}
